// EmbeddingBlock_42262478192812
// MI455X (gfx1250) — matrix operand order repaired and checked by reading, not compiled, not run on hardware
//
#include <hip/hip_runtime.h>

typedef __attribute__((ext_vector_type(16))) __bf16 v16bf;
typedef __attribute__((ext_vector_type(8)))  float  v8f;
typedef __attribute__((ext_vector_type(4)))  unsigned int v4u;
typedef __attribute__((ext_vector_type(8)))  int v8i;
typedef __attribute__((ext_vector_type(4)))  int v4i;

#define EMB 128

// Map W[k][n] into the per-lane B-fragment layout for v_wmma_f32_16x16x32_bf16.
// Fragment = 32 lanes x 16 halves (32B/lane). lanes 0-15: K 0..7,16..23 (N = lane&15),
// lanes 16-31: K 8..15,24..31.  fragIdx = noff*kTiles + koff.
__device__ __forceinline__ int pack_idx(int k, int n, int kTiles) {
  int noff = n >> 4;
  int koff = k >> 5;
  int kin  = k & 31;
  int lane = ((kin & 8) ? 16 : 0) | (n & 15);
  int h    = (kin & 7) | ((kin & 16) >> 1);
  return ((noff * kTiles + koff) * 32 + lane) * 16 + h;
}

// ---------------------------------------------------------------------------
// K1: fuse weights:  Wc = Wn*Wo1, Wd = Wn*Wo2, Ae = We*Wo3, Br = Wr*Wo4 (K-padded
// to 32 with zeros), bias_total = bn*(Wo1+Wo2) + be*Wo3 + br*Wo4 + bo.
// Results written pre-packed in WMMA B-fragment order (bf16).
// ---------------------------------------------------------------------------
__global__ __launch_bounds__(256) void k_weights(
    const float* __restrict__ Wn, const float* __restrict__ We,
    const float* __restrict__ Wr, const float* __restrict__ Wo,
    const float* __restrict__ bn, const float* __restrict__ be,
    const float* __restrict__ br, const float* __restrict__ bo,
    __bf16* __restrict__ wc, __bf16* __restrict__ wd,
    __bf16* __restrict__ ae, __bf16* __restrict__ brp,
    float* __restrict__ bias)
{
  int tid = blockIdx.x * blockDim.x + threadIdx.x;
  if (tid < 16384) {                        // Wc = Wn @ Wo[0:128]
    int a = tid >> 7, n = tid & 127;
    float s = 0.f;
    for (int m = 0; m < 128; ++m) s = fmaf(Wn[a * 128 + m], Wo[m * 128 + n], s);
    wc[pack_idx(a, n, 4)] = (__bf16)s;
  } else if (tid < 32768) {                 // Wd = Wn @ Wo[128:256]
    int r = tid - 16384; int a = r >> 7, n = r & 127;
    float s = 0.f;
    for (int m = 0; m < 128; ++m) s = fmaf(Wn[a * 128 + m], Wo[(128 + m) * 128 + n], s);
    wd[pack_idx(a, n, 4)] = (__bf16)s;
  } else if (tid < 49152) {                 // Ae = We @ Wo[256:384]
    int r = tid - 32768; int a = r >> 7, n = r & 127;
    float s = 0.f;
    for (int m = 0; m < 128; ++m) s = fmaf(We[a * 128 + m], Wo[(256 + m) * 128 + n], s);
    ae[pack_idx(a, n, 4)] = (__bf16)s;
  } else if (tid < 53248) {                 // Br = Wr @ Wo[384:512], K-padded 16->32
    int r = tid - 49152; int a = r >> 7, n = r & 127;
    float s = 0.f;
    if (a < 16)
      for (int m = 0; m < 128; ++m) s = fmaf(Wr[a * 128 + m], Wo[(384 + m) * 128 + n], s);
    brp[pack_idx(a, n, 1)] = (__bf16)s;
  } else if (tid < 53376) {                 // bias_total
    int n = tid - 53248;
    float s = bo[n];
    for (int m = 0; m < 128; ++m) {
      s = fmaf(bn[m], Wo[m * 128 + n] + Wo[(128 + m) * 128 + n], s);
      s = fmaf(be[m], Wo[(256 + m) * 128 + n], s);
    }
    for (int m = 0; m < 16; ++m) s = fmaf(br[m], Wo[(384 + m) * 128 + n], s);
    bias[n] = s;
  }
}

// Build the 16x32 bf16 A-fragment for this lane from a row-major f32 matrix.
// lanes 0-15: row = lane, halves = cols {k0..k0+7, k0+16..k0+23}
// lanes 16-31: row = lane-16, halves = cols {k0+8..k0+15, k0+24..k0+31}
__device__ __forceinline__ v16bf load_a_frag(const float* __restrict__ p) {
  float4 x0 = *(const float4*)(p);
  float4 x1 = *(const float4*)(p + 4);
  float4 x2 = *(const float4*)(p + 16);
  float4 x3 = *(const float4*)(p + 20);
  v16bf a;
  a[0]  = (__bf16)x0.x; a[1]  = (__bf16)x0.y; a[2]  = (__bf16)x0.z; a[3]  = (__bf16)x0.w;
  a[4]  = (__bf16)x1.x; a[5]  = (__bf16)x1.y; a[6]  = (__bf16)x1.z; a[7]  = (__bf16)x1.w;
  a[8]  = (__bf16)x2.x; a[9]  = (__bf16)x2.y; a[10] = (__bf16)x2.z; a[11] = (__bf16)x2.w;
  a[12] = (__bf16)x3.x; a[13] = (__bf16)x3.y; a[14] = (__bf16)x3.z; a[15] = (__bf16)x3.w;
  return a;
}

// ---------------------------------------------------------------------------
// K2: P = node_attr @ Wc, Q = node_attr @ Wd  (bf16 WMMA, f32 accum).
// One wave per 16-row node tile; A-fragments shared between P and Q.
// B-fragments streamed from global (32KB each, WGP$/L2 resident, reused 3125x).
// ---------------------------------------------------------------------------
__global__ __launch_bounds__(256) void k_nodeproj(
    const float* __restrict__ node_attr,
    const __bf16* __restrict__ wc, const __bf16* __restrict__ wd,
    float* __restrict__ P, float* __restrict__ Q, int nTiles)
{
  int wave = threadIdx.x >> 5, lane = threadIdx.x & 31;
  int tile = blockIdx.x * 8 + wave;
  if (tile >= nTiles) return;
  int half = lane >> 4, n0 = lane & 15;
  int base = tile * 16;

  const float* arow = node_attr + (size_t)(base + n0) * EMB + (half ? 8 : 0);
  v16bf aF[4];
#pragma unroll
  for (int koff = 0; koff < 4; ++koff)
    aF[koff] = load_a_frag(arow + koff * 32);

  const v16bf* fc = (const v16bf*)wc;
  const v16bf* fd = (const v16bf*)wd;
#pragma unroll
  for (int noff = 0; noff < 8; ++noff) {
    v8f cp = {};
    v8f cq = {};
#pragma unroll
    for (int koff = 0; koff < 4; ++koff) {
      v16bf bc = fc[(noff * 4 + koff) * 32 + lane];
      v16bf bd = fd[(noff * 4 + koff) * 32 + lane];
      cp = __builtin_amdgcn_wmma_f32_16x16x32_bf16(false, aF[koff], false, bc, (short)0, cp, false, false);
      cq = __builtin_amdgcn_wmma_f32_16x16x32_bf16(false, aF[koff], false, bd, (short)0, cq, false, false);
    }
#pragma unroll
    for (int r = 0; r < 8; ++r) {
      int nrow = base + r + (half ? 8 : 0);   // C/D layout: VGPR r -> M=r (+8 upper half)
      P[(size_t)nrow * EMB + noff * 16 + n0] = cp[r];
      Q[(size_t)nrow * EMB + noff * 16 + n0] = cq[r];
    }
  }
}

// ---------------------------------------------------------------------------
// K3: per-edge fused kernel (persistent, grid-stride over 16-edge tiles).
// out[e] = P[i(e)] + Q[j(e)] + edge_emb[e]@Ae + rbf[e]@Br + bias
// Packed weights (40KB) + bias staged in LDS once per block via the Tensor
// Data Mover (tensor_load_to_lds + s_wait_tensorcnt), issued by wave 0.
// ---------------------------------------------------------------------------
__global__ __launch_bounds__(256) void k_edge(
    const float* __restrict__ edge_emb, const float* __restrict__ rbf,
    const int* __restrict__ idnb_i, const int* __restrict__ idnb_j,
    const float* __restrict__ P, const float* __restrict__ Q,
    const unsigned char* __restrict__ wPacked,  // AeP(32KB)+BrP(8KB)+bias(512B)
    float* __restrict__ out, int nTiles)
{
  __shared__ __align__(16) unsigned char sBuf[41472];
  int t = threadIdx.x;

  if (t < 32) {
    // ---- TDM: DMA the 41472B packed-weight region into LDS (one 1D tile) ----
    // D# group0: count=1 | lds_addr | global_addr lo | global_addr hi + type=2
    unsigned long long ga = (unsigned long long)wPacked;
    unsigned int ldsAddr = (unsigned int)(size_t)(void*)sBuf;  // LDS aperture: low 32 bits = LDS offset
    v4u g0;
    g0[0] = 1u;                                   // count=1, is_restore=0, gather off
    g0[1] = ldsAddr;                              // lds_addr
    g0[2] = (unsigned int)ga;                     // global_addr[31:0]
    g0[3] = (unsigned int)((ga >> 32) & 0x01FFFFFFu) | (2u << 30);  // addr[56:32] | type=2
    // D# group1: data_size=4B, dim0 = tile0 = stride0 = 10368 dwords, dim1=tile1=1
    const unsigned int ND = 41472u / 4u;          // 10368 dwords
    v8i g1;
    g1[0] = (int)(2u << 16);                      // workgroup_mask=0, data_size=2 (4B)
    g1[1] = (int)((ND & 0xFFFFu) << 16);          // tensor_dim0[15:0]
    g1[2] = (int)((ND >> 16) | (1u << 16));       // tensor_dim0[31:16] | tensor_dim1[15:0]=1
    g1[3] = (int)((ND & 0xFFFFu) << 16);          // tensor_dim1[31:16]=0 | tile_dim0
    g1[4] = (int)1;                               // tile_dim1=1, tile_dim2=0
    g1[5] = (int)ND;                              // tensor_dim0_stride[31:0]
    g1[6] = (int)((ND & 0xFFFFu) << 16);          // stride0[47:32]=0 | stride1[15:0]
    g1[7] = 0;                                    // stride1[47:16]=0
    v4i z4 = {};                                  // groups 2/3 unused (<=2D)
    v8i z8 = {};
    __builtin_amdgcn_tensor_load_to_lds(g0, g1, z4, z4, z8, 0);
    __builtin_amdgcn_s_wait_tensorcnt(0);
  }
  __syncthreads();

  const v16bf* fE    = (const v16bf*)(sBuf);           // 32 frags (8 noff x 4 koff)
  const v16bf* fR    = (const v16bf*)(sBuf + 32768);   // 8 frags (noff), K padded
  const float* sBias = (const float*)(sBuf + 40960);   // 128 floats

  int wave = t >> 5, lane = t & 31;
  int half = lane >> 4, n0 = lane & 15;
  int stride = gridDim.x * 8;

  for (int tile = blockIdx.x * 8 + wave; tile < nTiles; tile += stride) {
    int eBase = tile * 16;

    // A-fragments: edge_emb rows (f32 -> bf16)
    const float* erow = edge_emb + (size_t)(eBase + n0) * EMB + (half ? 8 : 0);
    v16bf aE[4];
#pragma unroll
    for (int koff = 0; koff < 4; ++koff)
      aE[koff] = load_a_frag(erow + koff * 32);

    // rbf A-fragment: real K=16, halves 8..15 (K 16..31) are zero
    const float* rrow = rbf + (size_t)(eBase + n0) * 16 + (half ? 8 : 0);
    float4 r0 = *(const float4*)(rrow);
    float4 r1 = *(const float4*)(rrow + 4);
    v16bf aR;
    aR[0] = (__bf16)r0.x; aR[1] = (__bf16)r0.y; aR[2] = (__bf16)r0.z; aR[3] = (__bf16)r0.w;
    aR[4] = (__bf16)r1.x; aR[5] = (__bf16)r1.y; aR[6] = (__bf16)r1.z; aR[7] = (__bf16)r1.w;
#pragma unroll
    for (int h = 8; h < 16; ++h) aR[h] = (__bf16)0.0f;

    // prefetch next tile's edge rows (global_prefetch_b8)
    int nxt = tile + stride;
    if (nxt < nTiles)
      __builtin_prefetch(edge_emb + (size_t)(nxt * 16 + n0) * EMB, 0, 1);

    // init accumulators: bias + gathered P/Q rows (L2-resident)
    v8f acc[8];
#pragma unroll
    for (int r = 0; r < 8; ++r) {
      int e = eBase + r + (half ? 8 : 0);
      int ni = idnb_i[e];
      int nj = idnb_j[e];
      const float* Pr = P + (size_t)ni * EMB;
      const float* Qr = Q + (size_t)nj * EMB;
#pragma unroll
      for (int noff = 0; noff < 8; ++noff) {
        int c = noff * 16 + n0;
        acc[noff][r] = sBias[c] + Pr[c] + Qr[c];
      }
    }

    // GEMM: 4 K-steps for edge term + 1 (padded) for rbf term, per 16-col tile
#pragma unroll
    for (int noff = 0; noff < 8; ++noff) {
      v8f c = acc[noff];
#pragma unroll
      for (int koff = 0; koff < 4; ++koff) {
        v16bf b = fE[(noff * 4 + koff) * 32 + lane];
        c = __builtin_amdgcn_wmma_f32_16x16x32_bf16(false, aE[koff], false, b, (short)0, c, false, false);
      }
      v16bf b2 = fR[noff * 32 + lane];
      c = __builtin_amdgcn_wmma_f32_16x16x32_bf16(false, aR, false, b2, (short)0, c, false, false);
      acc[noff] = c;
    }

    // store
#pragma unroll
    for (int r = 0; r < 8; ++r) {
      int e = eBase + r + (half ? 8 : 0);
      float* orow = out + (size_t)e * EMB;
#pragma unroll
      for (int noff = 0; noff < 8; ++noff)
        orow[noff * 16 + n0] = acc[noff][r];
    }
  }
}

extern "C" void kernel_launch(void* const* d_in, const int* in_sizes, int n_in,
                              void* d_out, int out_size, void* d_ws, size_t ws_size,
                              hipStream_t stream) {
  const float* node_attr = (const float*)d_in[0];
  const float* edge_emb  = (const float*)d_in[1];
  const float* rbf       = (const float*)d_in[2];
  const int*   idnb_i    = (const int*)d_in[3];
  const int*   idnb_j    = (const int*)d_in[4];
  const float* Wn = (const float*)d_in[5];
  const float* bn = (const float*)d_in[6];
  const float* We = (const float*)d_in[7];
  const float* be = (const float*)d_in[8];
  const float* Wr = (const float*)d_in[9];
  const float* br = (const float*)d_in[10];
  const float* Wo = (const float*)d_in[11];
  const float* bo = (const float*)d_in[12];
  float* out = (float*)d_out;

  int nNodes = in_sizes[0] / EMB;   // 50000
  int nEdges = in_sizes[3];         // 800000

  // Workspace layout: P | Q | packed weights (AeP, BrP, bias contiguous for TDM)
  char* ws = (char*)d_ws;
  size_t pBytes = (size_t)nNodes * EMB * sizeof(float);
  float* P = (float*)ws;
  float* Q = (float*)(ws + pBytes);
  char* WB = ws + 2 * pBytes;
  __bf16* wc   = (__bf16*)(WB);            // 32768 B
  __bf16* wd   = (__bf16*)(WB + 32768);    // 32768 B
  __bf16* aew  = (__bf16*)(WB + 65536);    // 32768 B
  __bf16* brpk = (__bf16*)(WB + 98304);    //  8192 B
  float*  bias = (float*)(WB + 106496);    //   512 B

  k_weights<<<209, 256, 0, stream>>>(Wn, We, Wr, Wo, bn, be, br, bo,
                                     wc, wd, aew, brpk, bias);

  int nodeTiles = nNodes / 16;
  k_nodeproj<<<(nodeTiles + 7) / 8, 256, 0, stream>>>(node_attr, wc, wd, P, Q, nodeTiles);

  int edgeTiles = nEdges / 16;
  int blocks = (edgeTiles + 7) / 8;
  if (blocks > 2048) blocks = 2048;        // persistent blocks, grid-stride
  k_edge<<<blocks, 256, 0, stream>>>(edge_emb, rbf, idnb_i, idnb_j, P, Q,
                                     (const unsigned char*)(WB + 65536), out, edgeTiles);
}
